// GTN_19404662243717
// MI455X (gfx1250) — matrix operand order repaired and checked by reading, not compiled, not run on hardware
//
#include <hip/hip_runtime.h>
#include <math.h>

#define NNODES 50000
#define NEDGES 800000
#define FIN    128
#define HIDD   32
#define NH     4
#define NCLS   10
#define EPSF   1e-16f

typedef __attribute__((ext_vector_type(16))) _Float16 v16h;
typedef __attribute__((ext_vector_type(8)))  float    v8f;

// ---------------------------------------------------------------- utilities
__global__ void fill_f32(float* __restrict__ p, float v, int n) {
  int i = blockIdx.x * blockDim.x + threadIdx.x;
  if (i < n) p[i] = v;
}

__device__ inline void atomicMaxF(float* addr, float val) {
  // monotone bit-pattern trick: int-max for >=0, uint-min for <0
  if (val >= 0.0f) atomicMax((int*)addr, __float_as_int(val));
  else             atomicMin((unsigned int*)addr, __float_as_uint(val));
}

// ------------------------------------------------- WMMA GEMM (f32 in/out)
// C[M,NOUT] = A[M,K] @ W[K,NOUT] + b
// One wave per 16-row x (16*NT)-col output tile; K, NOUT compile-time so the
// K loop fully unrolls into back-to-back v_wmma ops sharing one A fragment.
// GUARD=true clamps+masks columns (only needed for NOUT=10 classifier).
template <int K, int NOUT, int NT, bool GUARD>
__global__ void __launch_bounds__(32) gemm_wmma(const float* __restrict__ A,
                                                const float* __restrict__ W,
                                                const float* __restrict__ b,
                                                float* __restrict__ C) {
  const int lane = threadIdx.x;
  const int half = lane >> 4;      // 0: lanes 0-15, 1: lanes 16-31
  const int l    = lane & 15;
  const int m0   = blockIdx.x * 16;
  const int n0   = blockIdx.y * (16 * NT);
  const int kbA  = half * 8;       // A: K 0-7/16-23 vs 8-15/24-31 per half
  const int kbB  = half * 8;       // B: K 0-7/16-23 vs 8-15/24-31 per half

  v8f acc[NT] = {};
  const float* arow = A + (size_t)(m0 + l) * K;

#pragma unroll
  for (int k0 = 0; k0 < K; k0 += 32) {
    v16h av;
#pragma unroll
    for (int j = 0; j < 8; ++j) av[j]     = (_Float16)arow[k0 + kbA + j];
#pragma unroll
    for (int j = 0; j < 8; ++j) av[8 + j] = (_Float16)arow[k0 + kbA + 16 + j];

#pragma unroll
    for (int t = 0; t < NT; ++t) {
      const int col  = n0 + t * 16 + l;
      // branchless: clamp address (always in-bounds), mask value afterwards
      const int colc = GUARD ? (col < NOUT ? col : NOUT - 1) : col;
      const bool cok = !GUARD || (col < NOUT);
      const float* wp = W + (size_t)(k0 + kbB) * NOUT + colc;
      v16h bv;
#pragma unroll
      for (int j = 0; j < 16; ++j) {
        const float wv = wp[(size_t)(j < 8 ? j : j + 8) * NOUT];   // unconditional load
        bv[j] = cok ? (_Float16)wv : (_Float16)0.0f;
      }
      acc[t] = __builtin_amdgcn_wmma_f32_16x16x32_f16(
          /*neg_a=*/false, av, /*neg_b=*/false, bv,
          /*c_mod=*/(short)0, acc[t], /*reuse_a=*/false, /*reuse_b=*/false);
    }
  }

#pragma unroll
  for (int t = 0; t < NT; ++t) {
    const int col = n0 + t * 16 + l;
    if (!GUARD || col < NOUT) {
      const float bias = b[col];
#pragma unroll
      for (int r = 0; r < 8; ++r) {
        const int row = m0 + r + half * 8;   // C/D layout: VGPR r -> M=r (+8)
        C[(size_t)row * NOUT + col] = acc[t][r] + bias;
      }
    }
  }
}

// ------------------------------------------ edge pass 1: logits + seg-max
__global__ void edge_logits_max(const float* __restrict__ q,
                                const float* __restrict__ kf,
                                const int* __restrict__ src,
                                const int* __restrict__ dst,
                                float* __restrict__ logits,
                                float* __restrict__ mx) {
  int t = blockIdx.x * blockDim.x + threadIdx.x;
  if (t >= NEDGES * NH) return;
  const int e = t >> 2;
  const int h = t & 3;
  const int s = src[e], d = dst[e];
  const float* qp = q  + (size_t)d * (NH * HIDD) + h * HIDD;
  const float* kp = kf + (size_t)s * (NH * HIDD) + h * HIDD;
  float acc = 0.0f;
#pragma unroll
  for (int c = 0; c < HIDD; ++c) acc += qp[c] * kp[c];
  const float logit = acc * 0.17677669529663687f;   // 1/sqrt(32)
  logits[t] = logit;
  atomicMaxF(&mx[(size_t)d * NH + h], logit);
}

// --------------------- edge pass 2: e = exp(logit-m); acc += e*v; den += e
__global__ void edge_accum(const float* __restrict__ v,
                           const int* __restrict__ src,
                           const int* __restrict__ dst,
                           const float* __restrict__ logits,
                           const float* __restrict__ mx,
                           float* __restrict__ den,
                           float* __restrict__ accb) {
  int t = blockIdx.x * blockDim.x + threadIdx.x;
  if (t >= NEDGES * NH * 8) return;
  const int part = t & 7;          // 8 threads per (edge,head), 4 chans each
  const int eh   = t >> 3;
  const int e = eh >> 2, h = eh & 3;
  const int s = src[e], d = dst[e];
  const float ev = expf(logits[eh] - mx[(size_t)d * NH + h]);
  if (part == 0) atomicAdd(&den[(size_t)d * NH + h], ev);
  const float* vp = v    + (size_t)s * (NH * HIDD) + h * HIDD + part * 4;
  float*       ap = accb + (size_t)d * (NH * HIDD) + h * HIDD + part * 4;
#pragma unroll
  for (int c = 0; c < 4; ++c) atomicAdd(&ap[c], ev * vp[c]);
}

// ----------------- node combine: head-mean of acc/(den+eps) + skip, ReLU
__global__ void node_combine(const float* __restrict__ accb,
                             const float* __restrict__ den,
                             const float* __restrict__ skip,
                             float* __restrict__ out) {
  int t = blockIdx.x * blockDim.x + threadIdx.x;
  if (t >= NNODES * HIDD) return;
  const int n = t >> 5;
  const int c = t & 31;
  float val = 0.0f;
#pragma unroll
  for (int h = 0; h < NH; ++h)
    val += accb[(size_t)n * (NH * HIDD) + h * HIDD + c] /
           (den[(size_t)n * NH + h] + EPSF);
  val = val * 0.25f + skip[t];
  out[t] = fmaxf(val, 0.0f);
}

// ---------------------------------------------------------------- launcher
static inline dim3 blk1d(long long n, int bs) { return dim3((unsigned)((n + bs - 1) / bs)); }

extern "C" void kernel_launch(void* const* d_in, const int* in_sizes, int n_in,
                              void* d_out, int out_size, void* d_ws, size_t ws_size,
                              hipStream_t stream) {
  const float* x   = (const float*)d_in[0];
  const int*   ei  = (const int*)  d_in[1];
  const float* q1w = (const float*)d_in[2];  const float* q1b = (const float*)d_in[3];
  const float* k1w = (const float*)d_in[4];  const float* k1b = (const float*)d_in[5];
  const float* v1w = (const float*)d_in[6];  const float* v1b = (const float*)d_in[7];
  const float* s1w = (const float*)d_in[8];  const float* s1b = (const float*)d_in[9];
  const float* q2w = (const float*)d_in[10]; const float* q2b = (const float*)d_in[11];
  const float* k2w = (const float*)d_in[12]; const float* k2b = (const float*)d_in[13];
  const float* v2w = (const float*)d_in[14]; const float* v2b = (const float*)d_in[15];
  const float* s2w = (const float*)d_in[16]; const float* s2b = (const float*)d_in[17];
  const float* lw  = (const float*)d_in[18]; const float* lb  = (const float*)d_in[19];

  const int* srcp = ei;            // edge_index[0]
  const int* dstp = ei + NEDGES;   // edge_index[1]

  // ---- workspace carve-up (256B aligned)
  char* ws = (char*)d_ws;
  size_t off = 0;
  auto carve = [&](size_t bytes) -> float* {
    float* p = (float*)(ws + off);
    off = (off + bytes + 255) & ~(size_t)255;
    return p;
  };
  float* qb   = carve((size_t)NNODES * FIN * 4);       // [N,128] q
  float* kb   = carve((size_t)NNODES * FIN * 4);       // [N,128] k
  float* vb   = carve((size_t)NNODES * FIN * 4);       // [N,128] v
  float* sb   = carve((size_t)NNODES * HIDD * 4);      // [N,32] skip
  float* mx   = carve((size_t)NNODES * NH * 4);        // [N,H] seg max
  float* den  = carve((size_t)NNODES * NH * 4);        // [N,H] seg sum
  float* accb = carve((size_t)NNODES * FIN * 4);       // [N,H,32] accum
  float* lg   = carve((size_t)NEDGES * NH * 4);        // [E,H] logits
  float* h1   = carve((size_t)NNODES * HIDD * 4);      // layer1 out
  float* h2   = carve((size_t)NNODES * HIDD * 4);      // layer2 out

  const dim3 wave(32);
  const unsigned MT = NNODES / 16;                     // 3125 row tiles

  // attention phase (shared by both layers): fills + 2 edge passes + combine
  auto attention = [&](float* outp, const float* skipb) {
    fill_f32<<<blk1d((long long)NNODES * NH, 256), 256, 0, stream>>>(
        mx, -INFINITY, NNODES * NH);
    fill_f32<<<blk1d((long long)NNODES * NH, 256), 256, 0, stream>>>(
        den, 0.0f, NNODES * NH);
    fill_f32<<<blk1d((long long)NNODES * FIN, 256), 256, 0, stream>>>(
        accb, 0.0f, NNODES * FIN);

    edge_logits_max<<<blk1d((long long)NEDGES * NH, 256), 256, 0, stream>>>(
        qb, kb, srcp, dstp, lg, mx);
    edge_accum<<<blk1d((long long)NEDGES * NH * 8, 256), 256, 0, stream>>>(
        vb, srcp, dstp, lg, mx, den, accb);
    node_combine<<<blk1d((long long)NNODES * HIDD, 256), 256, 0, stream>>>(
        accb, den, skipb, outp);
  };

  // ---- layer 1 (in = x, K = 128)
  gemm_wmma<FIN, FIN, 2, false><<<dim3(MT, 4), wave, 0, stream>>>(x, q1w, q1b, qb);
  gemm_wmma<FIN, FIN, 2, false><<<dim3(MT, 4), wave, 0, stream>>>(x, k1w, k1b, kb);
  gemm_wmma<FIN, FIN, 2, false><<<dim3(MT, 4), wave, 0, stream>>>(x, v1w, v1b, vb);
  gemm_wmma<FIN, HIDD, 2, false><<<dim3(MT, 1), wave, 0, stream>>>(x, s1w, s1b, sb);
  attention(h1, sb);

  // ---- layer 2 (in = h1, K = 32)
  gemm_wmma<HIDD, FIN, 2, false><<<dim3(MT, 4), wave, 0, stream>>>(h1, q2w, q2b, qb);
  gemm_wmma<HIDD, FIN, 2, false><<<dim3(MT, 4), wave, 0, stream>>>(h1, k2w, k2b, kb);
  gemm_wmma<HIDD, FIN, 2, false><<<dim3(MT, 4), wave, 0, stream>>>(h1, v2w, v2b, vb);
  gemm_wmma<HIDD, HIDD, 2, false><<<dim3(MT, 1), wave, 0, stream>>>(h1, s2w, s2b, sb);
  attention(h2, sb);

  // ---- final classifier: [N,32] @ [32,10] + b -> d_out (guarded columns)
  gemm_wmma<HIDD, NCLS, 1, true><<<dim3(MT, 1), wave, 0, stream>>>(
      h2, lw, lb, (float*)d_out);
}
